// CausalSelfAttention_11166914970610
// MI455X (gfx1250) — matrix operand order repaired and checked by reading, not compiled, not run on hardware
//
#include <hip/hip_runtime.h>

// ---------- types ----------
typedef __attribute__((ext_vector_type(16))) __bf16 v16bf;
typedef __attribute__((ext_vector_type(8)))  __bf16 v8bf;
typedef __attribute__((ext_vector_type(8)))  float  v8f;

#define T_SEQ 2048
#define C_DIM 1024
#define N_HEAD 16
#define D_HEAD 64
#define B_BATCH 2
#define M_ROWS (B_BATCH * T_SEQ)   // 4096

// ---------- helpers ----------
__device__ __forceinline__ unsigned short f32_to_bf16(float f) {
  unsigned int u = __float_as_uint(f);
  u += 0x7FFFu + ((u >> 16) & 1u);   // round-to-nearest-even
  return (unsigned short)(u >> 16);
}

__device__ __forceinline__ v8f wmma_bf16(v16bf a, v16bf b, v8f c) {
  return __builtin_amdgcn_wmma_f32_16x16x32_bf16(
      /*neg_a=*/false, a, /*neg_b=*/false, b,
      /*c_mod=*/(short)0, c, /*reuse_a=*/false, /*reuse_b=*/false);
}

// A-matrix 16x32 bf16 fragment: lane m = lane&15, half = lane>>4,
// elements j -> K = half*16 + j (contiguous, matches load_fragB)
__device__ __forceinline__ v16bf load_fragA(const unsigned short* src, int ld,
                                            int rowBase, int kBase, int lane) {
  int r = rowBase + (lane & 15);
  int half = (lane >> 4) & 1;
  const unsigned short* p = src + (size_t)r * ld + kBase + half * 16;
  union { v16bf v; v8bf h[2]; } u;
  u.h[0] = *(const v8bf*)(p);
  u.h[1] = *(const v8bf*)(p + 8);
  return u.v;
}

// B-matrix 32x16 bf16 fragment loaded from B^T (N x K row-major):
// lane n = lane&15, K block = 16*(lane>>4), 16 contiguous K values per lane.
__device__ __forceinline__ v16bf load_fragB(const unsigned short* src, int ld,
                                            int nBase, int kBase, int lane) {
  int n = nBase + (lane & 15);
  int kh = (lane >> 4) & 1;
  const unsigned short* p = src + (size_t)n * ld + kBase + kh * 16;
  union { v16bf v; v8bf h[2]; } u;
  u.h[0] = *(const v8bf*)(p);
  u.h[1] = *(const v8bf*)(p + 8);
  return u.v;
}

// Async stage of a contiguous block (nBytes, multiple of 128*16) global -> LDS.
// All 128 threads of the block participate; ends with s_wait_asynccnt; follow with a barrier.
__device__ __forceinline__ void async_stage_panel(const void* gsrc, void* lds,
                                                  int nBytes, int tid) {
  unsigned long long gbase = (unsigned long long)gsrc;
  unsigned int lbase = (unsigned int)(unsigned long long)lds;  // low 32 bits = LDS offset
  for (int c = tid * 16; c < nBytes; c += 128 * 16) {
    unsigned int laddr = lbase + c;
    unsigned long long gaddr = gbase + (unsigned long long)c;
    asm volatile("global_load_async_to_lds_b128 %0, %1, off"
                 :: "v"(laddr), "v"(gaddr) : "memory");
  }
  asm volatile("s_wait_asynccnt 0x0" ::: "memory");
}

// ---------- prep kernels ----------
__global__ __launch_bounds__(256) void cvt_x_kernel(const float* __restrict__ x,
                                                    unsigned short* __restrict__ xb, int n) {
  for (int i = blockIdx.x * blockDim.x + threadIdx.x; i < n; i += gridDim.x * blockDim.x)
    xb[i] = f32_to_bf16(x[i]);
}

__global__ __launch_bounds__(256) void transpose_w_kernel(const float* __restrict__ W,
                                                          unsigned short* __restrict__ Wt) {
  int idx = blockIdx.x * blockDim.x + threadIdx.x;   // over C*C
  int n = idx >> 10;          // output row (column of W)
  int k = idx & 1023;
  Wt[idx] = f32_to_bf16(W[(size_t)k * C_DIM + n]);
}

// ---------- QKV projection GEMM ----------
// Block (4 waves) shares one 16-row A panel staged async into LDS; each wave owns a
// 16x64 output tile. mode 0: out [B,H,T,D]; mode 1: out v^T [B,H,D,T]
__global__ __launch_bounds__(128) void proj_gemm_kernel(const unsigned short* __restrict__ A,
                                                        const unsigned short* __restrict__ Bt,
                                                        const float* __restrict__ bias,
                                                        unsigned short* __restrict__ out,
                                                        int mode) {
  __shared__ alignas(16) unsigned short Apanel[16 * C_DIM];   // 32 KB
  int tid = threadIdx.x;
  int lane = tid & 31;
  int wave = tid >> 5;
  int mTile = blockIdx.x >> 2;                 // 256 m tiles
  int g = blockIdx.x & 3;
  int mBase = mTile * 16;
  int nBase = g * 256 + wave * 64;

  // A panel: rows mBase..mBase+15, contiguous 32 KB in global (ld == C_DIM)
  async_stage_panel(A + (size_t)mBase * C_DIM, &Apanel[0], 16 * C_DIM * 2, tid);
  __syncthreads();
  const unsigned short* Ap = (const unsigned short*)Apanel;

  v8f acc0 = {}, acc1 = {}, acc2 = {}, acc3 = {};
  // software pipeline: preload k=0 fragments, prefetch k+32 inside the loop
  v16bf a  = load_fragA(Ap, C_DIM, 0, 0, lane);
  v16bf b0 = load_fragB(Bt, C_DIM, nBase +  0, 0, lane);
  v16bf b1 = load_fragB(Bt, C_DIM, nBase + 16, 0, lane);
  v16bf b2 = load_fragB(Bt, C_DIM, nBase + 32, 0, lane);
  v16bf b3 = load_fragB(Bt, C_DIM, nBase + 48, 0, lane);
  for (int k = 0; k < C_DIM; k += 32) {
    int kn = (k + 32) & (C_DIM - 1);           // wrap on last iter (redundant, harmless)
    __builtin_prefetch((const void*)(Bt + (size_t)(nBase + (lane & 15)) * C_DIM + kn + 128), 0, 0);
    v16bf an  = load_fragA(Ap, C_DIM, 0, kn, lane);
    v16bf nb0 = load_fragB(Bt, C_DIM, nBase +  0, kn, lane);
    v16bf nb1 = load_fragB(Bt, C_DIM, nBase + 16, kn, lane);
    v16bf nb2 = load_fragB(Bt, C_DIM, nBase + 32, kn, lane);
    v16bf nb3 = load_fragB(Bt, C_DIM, nBase + 48, kn, lane);
    acc0 = wmma_bf16(a, b0, acc0);
    acc1 = wmma_bf16(a, b1, acc1);
    acc2 = wmma_bf16(a, b2, acc2);
    acc3 = wmma_bf16(a, b3, acc3);
    a = an; b0 = nb0; b1 = nb1; b2 = nb2; b3 = nb3;
  }

  int half = lane >> 4;
  int nl = lane & 15;
  v8f accs[4] = {acc0, acc1, acc2, acc3};
#pragma unroll
  for (int j = 0; j < 4; ++j) {
    int n = nBase + j * 16 + nl;
    float bb = bias[n];
    int h = n >> 6, d = n & 63;
#pragma unroll
    for (int r = 0; r < 8; ++r) {
      int m = mBase + r + 8 * half;
      int b = m >> 11, t = m & 2047;
      float val = accs[j][r] + bb;
      size_t addr = (mode == 0)
        ? ((((size_t)b * N_HEAD + h) * T_SEQ + t) * D_HEAD + d)     // [B,H,T,D]
        : ((((size_t)b * N_HEAD + h) * D_HEAD + d) * T_SEQ + t);    // [B,H,D,T]
      out[addr] = f32_to_bf16(val);
    }
  }
}

// ---------- flash attention (one wave per (b,h,16-row Q block)) ----------
__global__ __launch_bounds__(32) void attention_kernel(const unsigned short* __restrict__ q,
                                                       const unsigned short* __restrict__ k,
                                                       const unsigned short* __restrict__ vT,
                                                       unsigned short* __restrict__ yb) {
  __shared__ alignas(16) unsigned short Pst[16 * 32];   // 1 KB staging for P tile

  int lane = threadIdx.x & 31;
  int half = lane >> 4;
  int nl = lane & 15;
  const int mTiles = T_SEQ / 16;          // 128
  int bh = blockIdx.x / mTiles;           // 0..31
  int mTile = blockIdx.x % mTiles;
  int b = bh >> 4, h = bh & 15;

  const unsigned short* qh = q  + (size_t)bh * T_SEQ * D_HEAD;
  const unsigned short* kh = k  + (size_t)bh * T_SEQ * D_HEAD;
  const unsigned short* vh = vT + (size_t)bh * D_HEAD * T_SEQ;

  v16bf qf0 = load_fragA(qh, D_HEAD, mTile * 16, 0, lane);
  v16bf qf1 = load_fragA(qh, D_HEAD, mTile * 16, 32, lane);

  v8f o0 = {}, o1 = {}, o2 = {}, o3 = {};
  float mrow[8], lsum[8];
#pragma unroll
  for (int r = 0; r < 8; ++r) { mrow[r] = -1e30f; lsum[r] = 0.0f; }

  const int sEnd = (mTile + 1) * 16;      // causal bound
  for (int s0 = 0; s0 < sEnd; s0 += 32) {
    // Issue ALL global fragment loads for this step up front: K tiles for the two
    // score WMMAs, and V tiles whose latency hides under the softmax VALU work.
    v16bf kb00 = load_fragB(kh, D_HEAD, s0,      0,  lane);
    v16bf kb01 = load_fragB(kh, D_HEAD, s0,      32, lane);
    v16bf kb10 = load_fragB(kh, D_HEAD, s0 + 16, 0,  lane);
    v16bf kb11 = load_fragB(kh, D_HEAD, s0 + 16, 32, lane);
    v16bf vb0  = load_fragB(vh, T_SEQ,  0, s0, lane);
    v16bf vb1  = load_fragB(vh, T_SEQ, 16, s0, lane);
    v16bf vb2  = load_fragB(vh, T_SEQ, 32, s0, lane);
    v16bf vb3  = load_fragB(vh, T_SEQ, 48, s0, lane);

    v8f st0 = {}, st1 = {};
    st0 = wmma_bf16(qf0, kb00, st0);
    st0 = wmma_bf16(qf1, kb01, st0);
    st1 = wmma_bf16(qf0, kb10, st1);
    st1 = wmma_bf16(qf1, kb11, st1);

    float pv[2][8];
    float tmax[8];
#pragma unroll
    for (int r = 0; r < 8; ++r) tmax[r] = -1e30f;
#pragma unroll
    for (int t = 0; t < 2; ++t) {
      int n = s0 + t * 16 + nl;
      bool colOk = ((n & 3) != 3);        // zero every STEP_DIM-th column
#pragma unroll
      for (int r = 0; r < 8; ++r) {
        int m = mTile * 16 + r + 8 * half;
        float val = (t ? st1[r] : st0[r]) * 0.125f;   // 1/sqrt(64)
        val = (colOk && (n <= m)) ? val : -1e30f;
        pv[t][r] = val;
        tmax[r] = fmaxf(tmax[r], val);
      }
    }

    // online softmax: row max across 16-lane half, rescale running state
#pragma unroll
    for (int r = 0; r < 8; ++r) {
      float v = tmax[r];
      v = fmaxf(v, __shfl_xor(v, 1));
      v = fmaxf(v, __shfl_xor(v, 2));
      v = fmaxf(v, __shfl_xor(v, 4));
      v = fmaxf(v, __shfl_xor(v, 8));
      float newm = fmaxf(mrow[r], v);
      float f = __expf(mrow[r] - newm);
      mrow[r] = newm;
      lsum[r] *= f;
      o0[r] *= f; o1[r] *= f; o2[r] *= f; o3[r] *= f;
    }

    // P = exp(S - m), stage 16x32 bf16 tile in LDS (accumulator -> A-fragment relayout)
#pragma unroll
    for (int t = 0; t < 2; ++t) {
#pragma unroll
      for (int r = 0; r < 8; ++r) {
        float p = __expf(pv[t][r] - mrow[r]);
        pv[t][r] = p;
        Pst[(r + 8 * half) * 32 + t * 16 + nl] = f32_to_bf16(p);
      }
    }
#pragma unroll
    for (int r = 0; r < 8; ++r) {
      float srow = pv[0][r] + pv[1][r];
      srow += __shfl_xor(srow, 1);
      srow += __shfl_xor(srow, 2);
      srow += __shfl_xor(srow, 4);
      srow += __shfl_xor(srow, 8);
      lsum[r] += srow;
    }

    asm volatile("s_wait_dscnt 0" ::: "memory");        // stores visible (in-wave, in-order DS)
    v16bf pf = load_fragA((const unsigned short*)Pst, 32, 0, 0, lane);

    o0 = wmma_bf16(pf, vb0, o0);
    o1 = wmma_bf16(pf, vb1, o1);
    o2 = wmma_bf16(pf, vb2, o2);
    o3 = wmma_bf16(pf, vb3, o3);
    asm volatile("s_wait_dscnt 0" ::: "memory");        // Pst loads done before next-iter stores
  }

  // normalize and write y as bf16 [B, T, C] (token-major for the output GEMM)
#pragma unroll
  for (int r = 0; r < 8; ++r) {
    float inv = 1.0f / lsum[r];
    int m = mTile * 16 + r + 8 * half;                  // token index within sequence
    size_t rowAddr = ((size_t)b * T_SEQ + m) * C_DIM + h * D_HEAD;
    yb[rowAddr +  0 + nl] = f32_to_bf16(o0[r] * inv);
    yb[rowAddr + 16 + nl] = f32_to_bf16(o1[r] * inv);
    yb[rowAddr + 32 + nl] = f32_to_bf16(o2[r] * inv);
    yb[rowAddr + 48 + nl] = f32_to_bf16(o3[r] * inv);
  }
}

// ---------- output projection GEMM (fp32 out) ----------
__global__ __launch_bounds__(128) void oproj_gemm_kernel(const unsigned short* __restrict__ A,
                                                         const unsigned short* __restrict__ Bt,
                                                         const float* __restrict__ bias,
                                                         float* __restrict__ out) {
  __shared__ alignas(16) unsigned short Apanel[16 * C_DIM];   // 32 KB
  int tid = threadIdx.x;
  int lane = tid & 31;
  int wave = tid >> 5;
  int mTile = blockIdx.x >> 2;
  int g = blockIdx.x & 3;
  int mBase = mTile * 16;
  int nBase = g * 256 + wave * 64;

  async_stage_panel(A + (size_t)mBase * C_DIM, &Apanel[0], 16 * C_DIM * 2, tid);
  __syncthreads();
  const unsigned short* Ap = (const unsigned short*)Apanel;

  v8f acc0 = {}, acc1 = {}, acc2 = {}, acc3 = {};
  v16bf a  = load_fragA(Ap, C_DIM, 0, 0, lane);
  v16bf b0 = load_fragB(Bt, C_DIM, nBase +  0, 0, lane);
  v16bf b1 = load_fragB(Bt, C_DIM, nBase + 16, 0, lane);
  v16bf b2 = load_fragB(Bt, C_DIM, nBase + 32, 0, lane);
  v16bf b3 = load_fragB(Bt, C_DIM, nBase + 48, 0, lane);
  for (int k = 0; k < C_DIM; k += 32) {
    int kn = (k + 32) & (C_DIM - 1);
    __builtin_prefetch((const void*)(Bt + (size_t)(nBase + (lane & 15)) * C_DIM + kn + 128), 0, 0);
    v16bf an  = load_fragA(Ap, C_DIM, 0, kn, lane);
    v16bf nb0 = load_fragB(Bt, C_DIM, nBase +  0, kn, lane);
    v16bf nb1 = load_fragB(Bt, C_DIM, nBase + 16, kn, lane);
    v16bf nb2 = load_fragB(Bt, C_DIM, nBase + 32, kn, lane);
    v16bf nb3 = load_fragB(Bt, C_DIM, nBase + 48, kn, lane);
    acc0 = wmma_bf16(a, b0, acc0);
    acc1 = wmma_bf16(a, b1, acc1);
    acc2 = wmma_bf16(a, b2, acc2);
    acc3 = wmma_bf16(a, b3, acc3);
    a = an; b0 = nb0; b1 = nb1; b2 = nb2; b3 = nb3;
  }

  int half = lane >> 4;
  int nl = lane & 15;
  v8f accs[4] = {acc0, acc1, acc2, acc3};
#pragma unroll
  for (int j = 0; j < 4; ++j) {
    int n = nBase + j * 16 + nl;
    float bb = bias[n];
#pragma unroll
    for (int r = 0; r < 8; ++r) {
      int m = mBase + r + 8 * half;
      out[(size_t)m * C_DIM + n] = accs[j][r] + bb;
    }
  }
}

// ---------- launch ----------
extern "C" void kernel_launch(void* const* d_in, const int* in_sizes, int n_in,
                              void* d_out, int out_size, void* d_ws, size_t ws_size,
                              hipStream_t stream) {
  (void)in_sizes; (void)n_in; (void)out_size; (void)ws_size;
  const float* x  = (const float*)d_in[0];
  const float* Wq = (const float*)d_in[1];
  const float* bq = (const float*)d_in[2];
  const float* Wk = (const float*)d_in[3];
  const float* bk = (const float*)d_in[4];
  const float* Wv = (const float*)d_in[5];
  const float* bv = (const float*)d_in[6];
  const float* Wp = (const float*)d_in[7];
  const float* bp = (const float*)d_in[8];

  unsigned short* w = (unsigned short*)d_ws;
  unsigned short* xb  = w;  w += (size_t)M_ROWS * C_DIM;          // 4M elems
  unsigned short* WqT = w;  w += (size_t)C_DIM * C_DIM;           // 1M
  unsigned short* WkT = w;  w += (size_t)C_DIM * C_DIM;
  unsigned short* WvT = w;  w += (size_t)C_DIM * C_DIM;
  unsigned short* WpT = w;  w += (size_t)C_DIM * C_DIM;
  unsigned short* qb  = w;  w += (size_t)B_BATCH * N_HEAD * T_SEQ * D_HEAD;  // 4M
  unsigned short* kb  = w;  w += (size_t)B_BATCH * N_HEAD * T_SEQ * D_HEAD;
  unsigned short* vT  = w;  w += (size_t)B_BATCH * N_HEAD * T_SEQ * D_HEAD;
  unsigned short* yb  = w;  w += (size_t)M_ROWS * C_DIM;

  // 1) precision conversion + weight transposes
  cvt_x_kernel<<<4096, 256, 0, stream>>>(x, xb, M_ROWS * C_DIM);
  transpose_w_kernel<<<4096, 256, 0, stream>>>(Wq, WqT);
  transpose_w_kernel<<<4096, 256, 0, stream>>>(Wk, WkT);
  transpose_w_kernel<<<4096, 256, 0, stream>>>(Wv, WvT);
  transpose_w_kernel<<<4096, 256, 0, stream>>>(Wp, WpT);

  // 2) QKV projections (v stored transposed per head)
  proj_gemm_kernel<<<1024, 128, 0, stream>>>(xb, WqT, bq, qb, 0);
  proj_gemm_kernel<<<1024, 128, 0, stream>>>(xb, WkT, bk, kb, 0);
  proj_gemm_kernel<<<1024, 128, 0, stream>>>(xb, WvT, bv, vT, 1);

  // 3) flash attention: 32 (b,h) pairs * 128 q-row tiles
  attention_kernel<<<B_BATCH * N_HEAD * (T_SEQ / 16), 32, 0, stream>>>(qb, kb, vT, yb);

  // 4) output projection -> fp32
  oproj_gemm_kernel<<<1024, 128, 0, stream>>>(yb, WpT, bp, (float*)d_out);
}
